// PhimoeSparseMoeBlock_35381940584952
// MI455X (gfx1250) — matrix operand order repaired and checked by reading, not compiled, not run on hardware
//
#include <hip/hip_runtime.h>
#include <hip/hip_bf16.h>
#include <math.h>

// Problem constants (fixed by setup_inputs): B=2,S=1024 -> T=2048, H=1024, F=2048, E=8
#define T_TOK 2048
#define H_DIM 1024
#define F_DIM 2048
#define N_EXP 8
#define EPS2  0.02f   // 2 * jitter_eps (jitter_eps = 0.01)

typedef unsigned short u16;
typedef unsigned int   u32;
typedef __attribute__((ext_vector_type(4)))  int    v4i;
typedef __attribute__((ext_vector_type(16))) __bf16 v16bf;
typedef __attribute__((ext_vector_type(8)))  float  v8f;

#define AS1 __attribute__((address_space(1)))
#define AS3 __attribute__((address_space(3)))

#if defined(__gfx1250__) && __has_builtin(__builtin_amdgcn_global_load_async_to_lds_b128)
#define HAS_ASYNC_LDS 1
#else
#define HAS_ASYNC_LDS 0
#endif

union FragU { u32 u[8]; v16bf v; };
static_assert(sizeof(FragU) == 32, "frag size");

__device__ __forceinline__ u16 f32_to_bf16(float f) {
  u32 u = __builtin_bit_cast(u32, f);
  u32 r = u + 0x7FFFu + ((u >> 16) & 1u);   // round-to-nearest-even
  return (u16)(r >> 16);
}
__device__ __forceinline__ u32 pack2_bf16(float lo, float hi) {
#if defined(__gfx1250__)
  // hardware packed convert, RNE (builtin not declared on this toolchain)
  u32 r;
  asm("v_cvt_pk_bf16_f32 %0, %1, %2" : "=v"(r) : "v"(lo), "v"(hi));
  return r;
#else
  return (u32)f32_to_bf16(lo) | ((u32)f32_to_bf16(hi) << 16);
#endif
}

__device__ __forceinline__ void wait_async_lds() {
#if HAS_ASYNC_LDS
#if __has_builtin(__builtin_amdgcn_s_wait_asynccnt)
  __builtin_amdgcn_s_wait_asynccnt(0);
#else
  asm volatile("s_wait_asynccnt 0" ::: "memory");
#endif
#endif
}

// A fragment: 16x32 bf16, row-major tile in LDS (stride in u16 elements).
// lane<16: M=lane, K=0..7 & 16..23 ; lane>=16: same M, K=8..15 & 24..31 (ISA 7.12.2)
template<int STRIDE>
__device__ __forceinline__ v16bf load_a_frag(const u16* sA, int mrow0, int lane) {
  const int m  = mrow0 + (lane & 15);
  const int kh = lane >> 4;
  FragU f;
#pragma unroll
  for (int i = 0; i < 8; ++i) {
    const int k = (i < 4) ? (kh * 8 + 2 * i) : (16 + kh * 8 + 2 * (i - 4));
    f.u[i] = *(const u32*)&sA[m * STRIDE + k];
  }
  return f.v;
}
// B fragment: 32x16 bf16 from an N-major [n][k] LDS tile.
template<int STRIDE>
__device__ __forceinline__ v16bf load_b_frag(const u16* sB, int nrow0, int lane) {
  const int n  = nrow0 + (lane & 15);
  const int kh = lane >> 4;
  FragU f;
#pragma unroll
  for (int i = 0; i < 8; ++i) {
    const int k = (i < 4) ? (kh * 8 + 2 * i) : (16 + kh * 8 + 2 * (i - 4));
    f.u[i] = *(const u32*)&sB[n * STRIDE + k];
  }
  return f.v;
}

// Copy one 16B chunk of a (already bf16) A tile row into LDS.
// Builtin prototype (from hipcc diagnostic): (int4 __device__*, int4 __shared__*, Iint, Iint)
__device__ __forceinline__ void copy_a_chunk(u16* lds_dst, const u16* gsrc) {
#if HAS_ASYNC_LDS
  __builtin_amdgcn_global_load_async_to_lds_b128(
      (AS1 v4i*)gsrc, (AS3 v4i*)lds_dst, 0, 0);
#else
  *(uint4*)lds_dst = *(const uint4*)gsrc;
#endif
}

// ---------------------------------------------------------------- utilities
__global__ void k_zero(float* __restrict__ p, int n) {
  int i = blockIdx.x * 256 + threadIdx.x;
  if (i < n) p[i] = 0.f;
}
__global__ void k_cvt_bf16(const float* __restrict__ x, u16* __restrict__ xb, int n) {
  int i = blockIdx.x * 256 + threadIdx.x;
  if (i < n) xb[i] = f32_to_bf16(x[i]);
}

// ------------------------------------------------- router + sparsemixer eval
__global__ __launch_bounds__(256) void k_router(
    const float* __restrict__ x, const float* __restrict__ rw,
    float* __restrict__ topk_w, int* __restrict__ topk_idx)
{
  const int t    = blockIdx.x;
  const int lane = threadIdx.x & 31;
  const int wave = threadIdx.x >> 5;            // 8 waves == 8 experts
  __shared__ float sl[N_EXP];

  const float* xp = x  + (size_t)t    * H_DIM;
  const float* wp = rw + (size_t)wave * H_DIM;
  float acc = 0.f;
  for (int j = lane; j < H_DIM; j += 32) acc += xp[j] * wp[j];
#pragma unroll
  for (int off = 16; off > 0; off >>= 1) acc += __shfl_xor(acc, off, 32);
  if (lane == 0) sl[wave] = acc;
  __syncthreads();

  if (threadIdx.x == 0) {
    float s[N_EXP];
#pragma unroll
    for (int e = 0; e < N_EXP; ++e) s[e] = sl[e];

    // ---- top-1 round ----
    int i1 = 0; float m1 = s[0];
#pragma unroll
    for (int e = 1; e < N_EXP; ++e) if (s[e] > m1) { m1 = s[e]; i1 = e; }
    float den1 = 0.f;
#pragma unroll
    for (int e = 0; e < N_EXP; ++e) {
      const bool msk = ((m1 - s[e]) / fmaxf(fabsf(s[e]), m1)) > EPS2;
      if (!msk) den1 += __expf(s[e] - m1);
    }
    const float w1 = __builtin_amdgcn_rcpf(den1);  // exp(m1-m1)/den1

    // ---- top-2 round (top-1 masked to -inf) ----
    int i2 = (i1 == 0) ? 1 : 0; float m2 = -INFINITY;
#pragma unroll
    for (int e = 0; e < N_EXP; ++e) {
      if (e == i1) continue;
      if (s[e] > m2) { m2 = s[e]; i2 = e; }
    }
    float den2 = 0.f;
#pragma unroll
    for (int e = 0; e < N_EXP; ++e) {
      if (e == i1) continue;                     // -inf in masked_scores
      const bool msk = ((m2 - s[e]) / fmaxf(fabsf(s[e]), m2)) > EPS2;
      if (!msk) den2 += __expf(s[e] - m2);
    }
    const float w2 = __builtin_amdgcn_rcpf(den2);

    topk_idx[2 * t + 0] = i1;  topk_idx[2 * t + 1] = i2;
    topk_w  [2 * t + 0] = w1;  topk_w  [2 * t + 1] = w2;
  }
}

// ------------------------------------------------- routing: count/scan/scatter
__global__ void k_count(const int* __restrict__ topk_idx, int* __restrict__ counts) {
  int t = blockIdx.x * 256 + threadIdx.x;
  if (t < T_TOK) {
    atomicAdd(&counts[topk_idx[2 * t + 0]], 1);
    atomicAdd(&counts[topk_idx[2 * t + 1]], 1);
  }
}
__global__ void k_scan(const int* __restrict__ counts, int* __restrict__ offsets,
                       int* __restrict__ cursor) {
  if (threadIdx.x == 0 && blockIdx.x == 0) {
    int o = 0;
    for (int e = 0; e < N_EXP; ++e) { offsets[e] = o; cursor[e] = o; o += counts[e]; }
    offsets[N_EXP] = o;
  }
}
__global__ void k_scatter(const int* __restrict__ topk_idx, const float* __restrict__ topk_w,
                          int* __restrict__ cursor, int* __restrict__ tok_of_row,
                          float* __restrict__ w_of_row) {
  int t = blockIdx.x * 256 + threadIdx.x;
  if (t < T_TOK) {
#pragma unroll
    for (int s = 0; s < 2; ++s) {
      const int e = topk_idx[2 * t + s];
      const int p = atomicAdd(&cursor[e], 1);
      tok_of_row[p] = t;
      w_of_row[p]   = topk_w[2 * t + s];
    }
  }
}

// ------------------------------------------------- GEMM1: inter = silu(xWg^T)*(xWu^T)
// grid: (F/64, maxMtiles=64, E), block 256 (8 waves; wave tile grid 2(M)x4(N))
__global__ __launch_bounds__(256) void k_gemm1(
    const float* __restrict__ wg, const float* __restrict__ wu,
    const u16* __restrict__ xb, const int* __restrict__ tok_of_row,
    const int* __restrict__ offsets, const int* __restrict__ counts,
    u16* __restrict__ inter)
{
  const int e    = blockIdx.z;
  const int cnt  = counts[e];
  const int row0 = blockIdx.y * 32;
  if (row0 >= cnt) return;                      // uniform early exit
  const int base  = offsets[e];
  const int ncol0 = blockIdx.x * 64;

  const int tid  = threadIdx.x;
  const int lane = tid & 31;
  const int wave = tid >> 5;
  const int wm   = wave & 1;                    // 0..1  (M sub-tile)
  const int wn   = wave >> 1;                   // 0..3  (N sub-tile)

  __shared__ u16 sA [32 * 40];                  // 32x32 bf16, 80B rows (16B aligned)
  __shared__ u16 sBg[64 * 36];
  __shared__ u16 sBu[64 * 36];
  __shared__ int sTok[32];

  if (tid < 32) {
    const int r = row0 + tid;
    sTok[tid] = tok_of_row[base + ((r < cnt) ? r : 0)];
  }
  __syncthreads();

  // loader roles
  const int rA  = (tid >> 2) & 31, kcA = (tid & 3) * 8;  // A: 16B chunk (tid<128)
  const int nB  = tid >> 2,        kqB = (tid & 3) * 8;  // B: 8 f32 per thread
  const u16*   aptr = &xb[(size_t)sTok[rA] * H_DIM + kcA];
  const float* gptr = &wg[((size_t)e * F_DIM + ncol0 + nB) * H_DIM + kqB];
  const float* uptr = &wu[((size_t)e * F_DIM + ncol0 + nB) * H_DIM + kqB];
  u16* aLds = &sA[rA * 40 + kcA];

  v8f accG = {}; v8f accU = {};

  // register-pipelined weight tiles
  float4 g0 = *(const float4*)(gptr);
  float4 g1 = *(const float4*)(gptr + 4);
  float4 u0 = *(const float4*)(uptr);
  float4 u1 = *(const float4*)(uptr + 4);

  for (int k0 = 0; k0 < H_DIM; k0 += 32) {
    // stage weight tiles (f32 -> bf16, v_cvt_pk_bf16_f32) into LDS
    *(u32*)&sBg[nB * 36 + kqB + 0] = pack2_bf16(g0.x, g0.y);
    *(u32*)&sBg[nB * 36 + kqB + 2] = pack2_bf16(g0.z, g0.w);
    *(u32*)&sBg[nB * 36 + kqB + 4] = pack2_bf16(g1.x, g1.y);
    *(u32*)&sBg[nB * 36 + kqB + 6] = pack2_bf16(g1.z, g1.w);
    *(u32*)&sBu[nB * 36 + kqB + 0] = pack2_bf16(u0.x, u0.y);
    *(u32*)&sBu[nB * 36 + kqB + 2] = pack2_bf16(u0.z, u0.w);
    *(u32*)&sBu[nB * 36 + kqB + 4] = pack2_bf16(u1.x, u1.y);
    *(u32*)&sBu[nB * 36 + kqB + 6] = pack2_bf16(u1.z, u1.w);

    // A tile: async copy straight to LDS (bf16, no conversion needed)
    if (tid < 128) copy_a_chunk(aLds, aptr + k0);
    wait_async_lds();
    __builtin_prefetch(gptr + k0 + 64, 0, 0);   // global_prefetch_b8 (L2 warm)
    __builtin_prefetch(uptr + k0 + 64, 0, 0);
    __syncthreads();

    // fetch next weight slice while WMMA consumes the current LDS tile
    if (k0 + 32 < H_DIM) {
      g0 = *(const float4*)(gptr + k0 + 32);
      g1 = *(const float4*)(gptr + k0 + 36);
      u0 = *(const float4*)(uptr + k0 + 32);
      u1 = *(const float4*)(uptr + k0 + 36);
    }

    const v16bf a  = load_a_frag<40>(sA,  wm * 16, lane);
    const v16bf bg = load_b_frag<36>(sBg, wn * 16, lane);
    const v16bf bu = load_b_frag<36>(sBu, wn * 16, lane);
    accG = __builtin_amdgcn_wmma_f32_16x16x32_bf16(false, a, false, bg, (short)0, accG, false, false);
    accU = __builtin_amdgcn_wmma_f32_16x16x32_bf16(false, a, false, bu, (short)0, accU, false, false);
    __syncthreads();
  }

  // epilogue: silu(G)*U -> bf16 intermediate (fast: exp + rcp)
  const int khC = lane >> 4, nC = lane & 15;
#pragma unroll
  for (int r = 0; r < 8; ++r) {
    const int mloc = wm * 16 + r + 8 * khC;     // C/D VGPR layout (ISA 7.12.2)
    const int grow = row0 + mloc;
    if (grow < cnt) {
      const float g = accG[r], u = accU[r];
      const float val = g * __builtin_amdgcn_rcpf(1.f + __expf(-g)) * u;
      inter[(size_t)(base + grow) * F_DIM + ncol0 + wn * 16 + nC] = f32_to_bf16(val);
    }
  }
}

// ------------------------------------------------- GEMM2: out += w * (inter @ Wd^T)
// grid: (H/64, maxMtiles=64, E), block 256
__global__ __launch_bounds__(256) void k_gemm2(
    const float* __restrict__ wd, const u16* __restrict__ inter,
    const int* __restrict__ tok_of_row, const float* __restrict__ w_of_row,
    const int* __restrict__ offsets, const int* __restrict__ counts,
    float* __restrict__ out)
{
  const int e    = blockIdx.z;
  const int cnt  = counts[e];
  const int row0 = blockIdx.y * 32;
  if (row0 >= cnt) return;
  const int base  = offsets[e];
  const int ncol0 = blockIdx.x * 64;

  const int tid  = threadIdx.x;
  const int lane = tid & 31;
  const int wave = tid >> 5;
  const int wm   = wave & 1;
  const int wn   = wave >> 1;

  __shared__ u16   sA[32 * 40];
  __shared__ u16   sB[64 * 36];
  __shared__ int   sTok[32];
  __shared__ float sW[32];

  if (tid < 32) {
    const int r  = row0 + tid;
    const int rr = (r < cnt) ? r : 0;
    sTok[tid] = tok_of_row[base + rr];
    sW[tid]   = w_of_row[base + rr];
  }
  __syncthreads();

  const int rA = (tid >> 2) & 31, kcA = (tid & 3) * 8;
  const int nB = tid >> 2,        kqB = (tid & 3) * 8;
  const int arow = base + (((row0 + rA) < cnt) ? (row0 + rA) : (cnt - 1));
  const u16*   aptr = &inter[(size_t)arow * F_DIM + kcA];
  const float* bptr = &wd[((size_t)e * H_DIM + ncol0 + nB) * F_DIM + kqB];
  u16* aLds = &sA[rA * 40 + kcA];

  v8f acc = {};

  float4 b0 = *(const float4*)(bptr);
  float4 b1 = *(const float4*)(bptr + 4);

  for (int k0 = 0; k0 < F_DIM; k0 += 32) {
    *(u32*)&sB[nB * 36 + kqB + 0] = pack2_bf16(b0.x, b0.y);
    *(u32*)&sB[nB * 36 + kqB + 2] = pack2_bf16(b0.z, b0.w);
    *(u32*)&sB[nB * 36 + kqB + 4] = pack2_bf16(b1.x, b1.y);
    *(u32*)&sB[nB * 36 + kqB + 6] = pack2_bf16(b1.z, b1.w);

    if (tid < 128) copy_a_chunk(aLds, aptr + k0);
    wait_async_lds();
    __builtin_prefetch(bptr + k0 + 64, 0, 0);
    __syncthreads();

    if (k0 + 32 < F_DIM) {
      b0 = *(const float4*)(bptr + k0 + 32);
      b1 = *(const float4*)(bptr + k0 + 36);
    }

    const v16bf a = load_a_frag<40>(sA, wm * 16, lane);
    const v16bf b = load_b_frag<36>(sB, wn * 16, lane);
    acc = __builtin_amdgcn_wmma_f32_16x16x32_bf16(false, a, false, b, (short)0, acc, false, false);
    __syncthreads();
  }

  const int khC = lane >> 4, nC = lane & 15;
#pragma unroll
  for (int r = 0; r < 8; ++r) {
    const int mloc = wm * 16 + r + 8 * khC;
    const int grow = row0 + mloc;
    if (grow < cnt) {
      const int   tok = sTok[mloc];
      const float w   = sW[mloc];
      atomicAdd(&out[(size_t)tok * H_DIM + ncol0 + wn * 16 + nC], w * acc[r]);
    }
  }
}

// ---------------------------------------------------------------- launch
extern "C" void kernel_launch(void* const* d_in, const int* in_sizes, int n_in,
                              void* d_out, int out_size, void* d_ws, size_t ws_size,
                              hipStream_t stream) {
  (void)in_sizes; (void)n_in; (void)out_size; (void)ws_size;
  const float* x  = (const float*)d_in[0];   // [2,1024,1024]
  const float* rw = (const float*)d_in[1];   // [8,1024]
  const float* wg = (const float*)d_in[2];   // [8,2048,1024]
  const float* wu = (const float*)d_in[3];   // [8,2048,1024]
  const float* wd = (const float*)d_in[4];   // [8,1024,2048]
  float* out = (float*)d_out;                // [2,1024,1024]

  // workspace layout (~21.1 MB)
  char* ws = (char*)d_ws;
  size_t off = 0;
  u16* xb         = (u16*)(ws + off);  off += (size_t)T_TOK * H_DIM * 2;       // 4 MiB
  u16* inter      = (u16*)(ws + off);  off += (size_t)2 * T_TOK * F_DIM * 2;   // 16 MiB
  float* topk_w   = (float*)(ws + off); off += (size_t)T_TOK * 2 * 4;
  float* w_of_row = (float*)(ws + off); off += (size_t)2 * T_TOK * 4;
  int* topk_idx   = (int*)(ws + off);  off += (size_t)T_TOK * 2 * 4;
  int* tok_of_row = (int*)(ws + off);  off += (size_t)2 * T_TOK * 4;
  int* counts     = (int*)(ws + off);  off += (size_t)N_EXP * 4;
  int* offsets    = (int*)(ws + off);  off += (size_t)(N_EXP + 1) * 4;
  int* cursor     = (int*)(ws + off);  off += (size_t)N_EXP * 4;

  const int nOut = T_TOK * H_DIM;

  k_zero<<<(nOut + 255) / 256, 256, 0, stream>>>(out, nOut);
  k_zero<<<1, 256, 0, stream>>>((float*)counts, N_EXP);          // bit-zero ints
  k_cvt_bf16<<<(nOut + 255) / 256, 256, 0, stream>>>(x, xb, nOut);

  k_router<<<T_TOK, 256, 0, stream>>>(x, rw, topk_w, topk_idx);
  k_count<<<(T_TOK + 255) / 256, 256, 0, stream>>>(topk_idx, counts);
  k_scan<<<1, 64, 0, stream>>>(counts, offsets, cursor);
  k_scatter<<<(T_TOK + 255) / 256, 256, 0, stream>>>(topk_idx, topk_w, cursor,
                                                     tok_of_row, w_of_row);

  // grouped GEMMs: fixed worst-case grid, uniform early exit per expert tile
  k_gemm1<<<dim3(F_DIM / 64, T_TOK / 32, N_EXP), 256, 0, stream>>>(
      wg, wu, xb, tok_of_row, offsets, counts, inter);
  k_gemm2<<<dim3(H_DIM / 64, T_TOK / 32, N_EXP), 256, 0, stream>>>(
      wd, inter, tok_of_row, w_of_row, offsets, counts, out);
}
